// PointNet_21345987461166
// MI455X (gfx1250) — matrix operand order repaired and checked by reading, not compiled, not run on hardware
//
#include <hip/hip_runtime.h>
#include <hip/hip_bf16.h>

typedef __attribute__((ext_vector_type(16))) _Float16 v16h;
typedef __attribute__((ext_vector_type(8)))  float    v8f;
typedef __attribute__((ext_vector_type(2)))  __fp16   fp16x2;

#define CIN   64
#define CREAL 67        // 64 feats + 3 rel-pos
#define CB    68        // +1 bias channel (==1.0, W1 col 67 = b1)
#define CPAD  96        // 3 K-chunks of 32
#define HID   64
#define OUTC  128
#define KNB   16
#define WPB   8         // 8 wave32 per block
#define NPTS  65536     // N (hardcoded: enables immediate-offset gather)
#define MPTS  65536     // M

union Frag { v16h v; uint4 q[2]; };

__device__ __forceinline__ unsigned pk2(float a, float b) {
    fp16x2 t = __builtin_amdgcn_cvt_pkrtz(a, b);
    return __builtin_bit_cast(unsigned, t);
}

__global__ __launch_bounds__(256)
void pointconv_wmma_kernel(const float* __restrict__ x,
                           const float* __restrict__ pos,
                           const float* __restrict__ sup,
                           const float* __restrict__ W1,
                           const float* __restrict__ b1,
                           const float* __restrict__ W2,
                           const float* __restrict__ b2,
                           const long long* __restrict__ ind,
                           float* __restrict__ out)
{
    __shared__ __align__(16) _Float16 w1s[HID * CPAD];       // W1|b1, zero-padded
    __shared__ __align__(16) _Float16 w2s[OUTC * HID];
    __shared__ __align__(16) _Float16 tin[WPB][KNB * CPAD];  // per-wave input tile
    __shared__ __align__(16) _Float16 th1[WPB][KNB * HID];   // per-wave hidden tile

    const int tid = threadIdx.x;

    // ---- cooperative weight preload (branch-free, fp32 -> f16) ----
    for (int i = tid; i < HID * CREAL; i += 256) {           // W1 body
        int h = i / CREAL, c = i - h * CREAL;
        w1s[h * CPAD + c] = (_Float16)W1[i];
    }
    for (int h = tid; h < HID; h += 256)                     // bias channel
        w1s[h * CPAD + CREAL] = (_Float16)b1[h];
    for (int i = tid; i < HID * (CPAD - CB); i += 256) {     // zero pad 68..95
        int h = i / (CPAD - CB), c = CB + (i - h * (CPAD - CB));
        w1s[h * CPAD + c] = (_Float16)0.f;
    }
    for (int i = tid; i < OUTC * HID; i += 256)
        w2s[i] = (_Float16)W2[i];
    __syncthreads();

    const int lane = tid & 31;
    const int wave = tid >> 5;
    const int col  = lane & 15;
    const int hi   = lane >> 4;
    _Float16* my_in = tin[wave];
    _Float16* my_h1 = th1[wave];

    // ---- W2^T B-fragments: register-resident for the whole kernel ----
    // B layout: lane owns column (output ch o = r2*16+col), K = hi*8 + 0..7, 16 + hi*8 + 0..7
    Frag w2f[16];
    #pragma unroll
    for (int r2 = 0; r2 < 8; ++r2)
        #pragma unroll
        for (int ck = 0; ck < 2; ++ck) {
            const _Float16* p = w2s + (r2 * 16 + col) * HID + ck * 32 + hi * 8;
            w2f[r2 * 2 + ck].q[0] = *(const uint4*)(p);
            w2f[r2 * 2 + ck].q[1] = *(const uint4*)(p + 16);
        }
    float b2v[8];
    #pragma unroll
    for (int r2 = 0; r2 < 8; ++r2) b2v[r2] = b2[r2 * 16 + col];

    // ---- init constant pad region of the input tile once (gather never writes it) ----
    if (hi == 0) {
        my_in[col * CPAD + CREAL] = (_Float16)1.0f;          // bias channel
        #pragma unroll
        for (int c = CB; c < CPAD; c += 2)
            *(unsigned*)&my_in[col * CPAD + c] = 0u;
    }

    const int wavesTotal = (int)gridDim.x * WPB;
    for (int m = (int)blockIdx.x * WPB + wave; m < MPTS; m += wavesTotal) {

        // ---- gather 16 neighbors x 67 ch -> LDS tile [neighbor][channel] f16 ----
        const int gi = (int)ind[(long long)m * KNB + col];
        const float* xc = x + (size_t)(hi * 32) * NPTS + gi; // imm offsets c*256KB
        float f[32];
        #pragma unroll
        for (int c = 0; c < 32; ++c) f[c] = xc[(size_t)c * NPTS];
        #pragma unroll
        for (int g4 = 0; g4 < 4; ++g4) {                     // 8 ch per b128 store
            uint4 P;
            P.x = pk2(f[g4 * 8 + 0], f[g4 * 8 + 1]);
            P.y = pk2(f[g4 * 8 + 2], f[g4 * 8 + 3]);
            P.z = pk2(f[g4 * 8 + 4], f[g4 * 8 + 5]);
            P.w = pk2(f[g4 * 8 + 6], f[g4 * 8 + 7]);
            *(uint4*)&my_in[col * CPAD + hi * 32 + g4 * 8] = P;
        }
        if (hi == 0) {                                       // rel-pos channels 64..66
            float d0 = pos[(size_t)0 * NPTS + gi] - sup[(size_t)0 * MPTS + m];
            float d1 = pos[(size_t)1 * NPTS + gi] - sup[(size_t)1 * MPTS + m];
            float d2 = pos[(size_t)2 * NPTS + gi] - sup[(size_t)2 * MPTS + m];
            *(unsigned*)&my_in[col * CPAD + 64] = pk2(d0, d1);
            my_in[col * CPAD + 66] = (_Float16)d2;
        }
        asm volatile("s_wait_dscnt 0" ::: "memory");

        // ---- stage 1: h1 = relu(W1ext * in)  (B frags reused across 4 row tiles) ----
        Frag Bf[3];
        #pragma unroll
        for (int ck = 0; ck < 3; ++ck) {
            const _Float16* p = my_in + col * CPAD + ck * 32 + hi * 8;
            Bf[ck].q[0] = *(const uint4*)(p);
            Bf[ck].q[1] = *(const uint4*)(p + 16);
        }
        #pragma unroll
        for (int r = 0; r < 4; ++r) {
            v8f acc = {};
            #pragma unroll
            for (int ck = 0; ck < 3; ++ck) {
                Frag A;                                       // W1 A-frag from LDS
                const _Float16* p = w1s + (r * 16 + col) * CPAD + ck * 32 + hi * 8;
                A.q[0] = *(const uint4*)(p);
                A.q[1] = *(const uint4*)(p + 16);
                acc = __builtin_amdgcn_wmma_f32_16x16x32_f16(
                          false, A.v, false, Bf[ck].v, (short)0, acc, false, false);
            }
            uint4 H;                                          // relu + pack 8 rows
            H.x = pk2(fmaxf(acc[0], 0.f), fmaxf(acc[1], 0.f));
            H.y = pk2(fmaxf(acc[2], 0.f), fmaxf(acc[3], 0.f));
            H.z = pk2(fmaxf(acc[4], 0.f), fmaxf(acc[5], 0.f));
            H.w = pk2(fmaxf(acc[6], 0.f), fmaxf(acc[7], 0.f));
            *(uint4*)&my_h1[col * HID + r * 16 + hi * 8] = H;
        }
        asm volatile("s_wait_dscnt 0" ::: "memory");

        // ---- stage 2 (transposed): D[k][o] = h1^T x W2^T; pool = in-lane max ----
        Frag Af[2];                                           // h1^T A-frags, reused x8
        #pragma unroll
        for (int ck = 0; ck < 2; ++ck) {
            const _Float16* p = my_h1 + col * HID + ck * 32 + hi * 8;
            Af[ck].q[0] = *(const uint4*)(p);
            Af[ck].q[1] = *(const uint4*)(p + 16);
        }
        #pragma unroll
        for (int r2 = 0; r2 < 8; ++r2) {
            v8f acc = {};
            #pragma unroll
            for (int ck = 0; ck < 2; ++ck)
                acc = __builtin_amdgcn_wmma_f32_16x16x32_f16(
                          false, Af[ck].v, false, w2f[r2 * 2 + ck].v,
                          (short)0, acc, false, false);
            // rows = neighbors: reduce 8 in-lane + 1 exchange across lane halves
            float mx = fmaxf(acc[0], acc[1]);
            mx = fmaxf(mx, fmaxf(acc[2], acc[3]));
            mx = fmaxf(mx, fmaxf(acc[4], acc[5]));
            mx = fmaxf(mx, fmaxf(acc[6], acc[7]));
            mx = fmaxf(mx, __shfl_xor(mx, 16, 32));
            if (hi == (r2 & 1))                               // balance stores
                out[(size_t)(r2 * 16 + col) * MPTS + m] = mx + b2v[r2];
        }
    }
}

extern "C" void kernel_launch(void* const* d_in, const int* in_sizes, int n_in,
                              void* d_out, int out_size, void* d_ws, size_t ws_size,
                              hipStream_t stream) {
    const float*     x   = (const float*)d_in[0];
    const float*     pos = (const float*)d_in[1];
    const float*     sup = (const float*)d_in[2];
    const float*     W1  = (const float*)d_in[3];
    const float*     b1  = (const float*)d_in[4];
    const float*     W2  = (const float*)d_in[5];
    const float*     b2  = (const float*)d_in[6];
    const long long* ind = (const long long*)d_in[7];
    float* out = (float*)d_out;

    // M = 65536, 8 waves/block, grid-stride: 4 points per wave
    pointconv_wmma_kernel<<<2048, 256, 0, stream>>>(
        x, pos, sup, W1, b1, W2, b2, ind, out);
}
